// KNNLearner_18013092839500
// MI455X (gfx1250) — matrix operand order repaired and checked by reading, not compiled, not run on hardware
//
#include <hip/hip_runtime.h>
#include <climits>

#define NS_   8192
#define NQ_   4096
#define DIM_  1024
#define NCLS_ 64
#define NCHUNK 8
#define CHUNK  (NS_ / NCHUNK)   // 1024 supports per chunk
#define TQ 128                  // query tile per block
#define TS 128                  // support tile per iteration
#define KC 32                   // K step (bf16 WMMA K)
#define KP 16                   // K-pairs (u32) per K step
#define LDSP 20                 // padded LDS row stride in u32 (= 16 data + 4 pad)
#define KPROW (DIM_ / 2)        // 512 packed u32 per row in global packed arrays

#if __has_builtin(__builtin_amdgcn_tensor_load_to_lds) && \
    __has_builtin(__builtin_amdgcn_s_wait_tensorcnt)
#define USE_TDM 1
#else
#define USE_TDM 0
#endif

typedef __bf16   v16bf __attribute__((ext_vector_type(16)));
typedef float    v8f   __attribute__((ext_vector_type(8)));
typedef unsigned v4u   __attribute__((ext_vector_type(4)));
typedef int      v4i   __attribute__((ext_vector_type(4)));
typedef int      v8i_t __attribute__((ext_vector_type(8)));

union Frag { unsigned u[8]; v16bf v; };

__device__ __forceinline__ void set8(Frag& f, uint4 a, uint4 b) {
  f.u[0] = a.x; f.u[1] = a.y; f.u[2] = a.z; f.u[3] = a.w;
  f.u[4] = b.x; f.u[5] = b.y; f.u[6] = b.z; f.u[7] = b.w;
}

// Split an f32 pair into packed bf16 hi (exact truncation) and bf16 lo
// (truncated residual, |err| <= 2^-17 |x|). Element K even -> bits[15:0].
__device__ __forceinline__ void split_pack(float xa, float xb, unsigned &hi, unsigned &lo) {
  unsigned bx = __float_as_uint(xa);
  unsigned by = __float_as_uint(xb);
  unsigned hx = bx & 0xFFFF0000u;
  unsigned hy = by & 0xFFFF0000u;
  float rx = xa - __uint_as_float(hx);
  float ry = xb - __uint_as_float(hy);
  hi = (hx >> 16) | hy;
  lo = (__float_as_uint(rx) >> 16) | (__float_as_uint(ry) & 0xFFFF0000u);
}

#if USE_TDM
// One TDM descriptor: 128 rows x 16 u32 tile from a 512-u32-stride tensor into LDS,
// padding +4 DWORDs every 16 DWORDs -> LDS row stride 20 u32 (bank-conflict free).
__device__ __forceinline__ void tdm_load_tile(const unsigned* g, unsigned ldsoff) {
  unsigned long long ga = (unsigned long long)g;
  v4u g0;
  g0[0] = 1u;                                             // count=1, no gather
  g0[1] = ldsoff;                                         // lds_addr (bytes)
  g0[2] = (unsigned)ga;                                   // global_addr[31:0]
  g0[3] = ((unsigned)(ga >> 32) & 0x01FFFFFFu)            // global_addr[56:32]
          | 0x80000000u;                                  // type=2 ("image")
  v8i_t g1;
  g1[0] = (2 << 16)                                       // data_size = 4B
        | (1 << 20)                                       // pad_enable
        | (3 << 22)                                       // pad_interval = 16 DWORDs
        | (3 << 25);                                      // pad_amount   = 4 DWORDs
  g1[1] = (int)(512u << 16);                              // tensor_dim0 = 512 (lo16)
  g1[2] = (int)(128u << 16);                              // tensor_dim1 = 128 (lo16)
  g1[3] = (int)(16u << 16);                               // tile_dim0 = 16 u32
  g1[4] = 128;                                            // tile_dim1 = 128 rows
  g1[5] = 512;                                            // tensor_dim0_stride = 512
  g1[6] = 0;
  g1[7] = 0;
  v4i z4 = {0, 0, 0, 0};
#if __clang_major__ >= 23
  v8i_t z8 = {0, 0, 0, 0, 0, 0, 0, 0};
  __builtin_amdgcn_tensor_load_to_lds(g0, g1, z4, z4, z8, 0);
#else
  __builtin_amdgcn_tensor_load_to_lds(g0, g1, z4, z4, 0);
#endif
}
#endif

// ---------------- Kernel 0: support norms + labels ----------------
__global__ __launch_bounds__(256)
void knn_prep(const float* __restrict__ Sg, const float* __restrict__ oh,
              float* __restrict__ s2g, int* __restrict__ slab) {
  const int wave = threadIdx.x >> 5;
  const int lane = threadIdx.x & 31;
  const int row  = blockIdx.x * 8 + wave;      // one wave per support row
  const float* sp = Sg + (size_t)row * DIM_;
  float s = 0.f;
#pragma unroll
  for (int i = 0; i < 8; ++i) {
    float4 v = *(const float4*)&sp[lane * 4 + i * 128];
    s += v.x * v.x + v.y * v.y + v.z * v.z + v.w * v.w;
  }
  int lab = -1;
  if (lane < 16) {
    float4 v = *(const float4*)&oh[(size_t)row * NCLS_ + lane * 4];
    if (v.x > 0.5f) lab = lane * 4 + 0;
    if (v.y > 0.5f) lab = lane * 4 + 1;
    if (v.z > 0.5f) lab = lane * 4 + 2;
    if (v.w > 0.5f) lab = lane * 4 + 3;
  }
#pragma unroll
  for (int off = 16; off; off >>= 1) {
    s += __shfl_xor(s, off, 32);
    int ol = __shfl_xor(lab, off, 32);
    lab = (ol > lab) ? ol : lab;
  }
  if (lane == 0) { s2g[row] = s; slab[row] = lab; }
}

// ---------------- Kernel 0b: fp32 -> packed bf16 hi/lo split ----------------
__global__ __launch_bounds__(256)
void knn_split(const float* __restrict__ src, unsigned* __restrict__ dhi,
               unsigned* __restrict__ dlo, int n4) {
  int t = blockIdx.x * 256 + threadIdx.x;
  if (t >= n4) return;
  float4 f = ((const float4*)src)[t];
  unsigned h0, l0, h1, l1;
  split_pack(f.x, f.y, h0, l0);
  split_pack(f.z, f.w, h1, l1);
  ((uint2*)dhi)[t] = make_uint2(h0, h1);
  ((uint2*)dlo)[t] = make_uint2(l0, l1);
}

// ---------------- Kernel 1: split-bf16 WMMA GEMM + running argmin ----------------
// MODE 0: stage from f32, converting once per element at LDS-store time.
// MODE 1: packed arrays in global; staging is a straight b128 copy.
// MODE 2: packed arrays in global; TDM (tensor_load_to_lds) double-buffered staging.
template <int MODE>
__global__ __launch_bounds__(256)
void knn_gemm_argmin(const float* __restrict__ Sg, const float* __restrict__ Qg,
                     const uint4* __restrict__ Qh4, const uint4* __restrict__ Ql4,
                     const uint4* __restrict__ Sh4, const uint4* __restrict__ Sl4,
                     const float* __restrict__ s2g,
                     float* __restrict__ pv, int* __restrict__ pi) {
  __shared__ __align__(16) unsigned Qhp[2][TQ * LDSP];
  __shared__ __align__(16) unsigned Qlp[2][TQ * LDSP];
  __shared__ __align__(16) unsigned Shp[2][TS * LDSP];
  __shared__ __align__(16) unsigned Slp[2][TS * LDSP];
  __shared__ float S2s[CHUNK];
  __shared__ float RedV[TQ][32];
  __shared__ int   RedI[TQ][32];

  const int tid    = threadIdx.x;
  const int wave   = tid >> 5;
  const int lane   = tid & 31;
  const int lanelo = lane & 15;
  const int h      = lane >> 4;       // lane half
  const int waveM  = wave >> 1;       // 0..3 : 32 query rows each
  const int waveN  = wave & 1;        // 0..1 : 64 support cols each
  const int qbase  = blockIdx.x * TQ;
  const int chunk  = blockIdx.y;
  const int sbase0 = chunk * CHUNK;

  const unsigned* Qh = (const unsigned*)Qh4;
  const unsigned* Ql = (const unsigned*)Ql4;
  const unsigned* Sh = (const unsigned*)Sh4;
  const unsigned* Sl = (const unsigned*)Sl4;

  for (int i = tid; i < CHUNK; i += 256) S2s[i] = s2g[sbase0 + i];
  __syncthreads();

  float rv[2][8];
  int   ri[2][8];
#pragma unroll
  for (int m = 0; m < 2; ++m)
#pragma unroll
    for (int r = 0; r < 8; ++r) { rv[m][r] = 3.0e38f; ri[m][r] = 0; }

  const v8f zacc = {0.f, 0.f, 0.f, 0.f, 0.f, 0.f, 0.f, 0.f};
  const int GSTEPS = (CHUNK / TS) * (DIM_ / KC);          // 256

#if USE_TDM
  auto tdm_issue = [&](int g) {
    if (MODE == 2 && wave < 4) {
      int stn = g >> 5;                                    // s-tile of step g
      int kp  = (g & 31) * KP;                             // k-pair column base
      int buf = g & 1;
      size_t qoff = (size_t)qbase * KPROW + kp;
      size_t soff = (size_t)(sbase0 + stn * TS) * KPROW + kp;
      if      (wave == 0) tdm_load_tile(Qh + qoff, (unsigned)(size_t)&Qhp[buf][0]);
      else if (wave == 1) tdm_load_tile(Ql + qoff, (unsigned)(size_t)&Qlp[buf][0]);
      else if (wave == 2) tdm_load_tile(Sh + soff, (unsigned)(size_t)&Shp[buf][0]);
      else                tdm_load_tile(Sl + soff, (unsigned)(size_t)&Slp[buf][0]);
    }
  };
  if (MODE == 2) tdm_issue(0);
#endif

  for (int st = 0; st < CHUNK / TS; ++st) {
    const int sbase = sbase0 + st * TS;
    v8f acc[2][4];
#pragma unroll
    for (int m = 0; m < 2; ++m)
#pragma unroll
      for (int n = 0; n < 4; ++n) acc[m][n] = zacc;

    for (int kstep = 0; kstep < DIM_ / KC; ++kstep) {
      const int g   = st * (DIM_ / KC) + kstep;
      const int buf = (MODE == 2) ? (g & 1) : 0;
      const int k0  = kstep * KC;

      // ---- stage packed bf16 hi/lo tiles into LDS ----
      if (MODE == 0) {
#pragma unroll
        for (int i = 0; i < 4; ++i) {
          int linear = tid + i * 256;                      // 0..1023
          int row = linear >> 3;                           // 0..127
          int kk  = (linear & 7) * 4;                      // f32 col 0..28
          const float* qp = &Qg[(size_t)(qbase + row) * DIM_ + k0 + kk];
          const float* sp = &Sg[(size_t)(sbase + row) * DIM_ + k0 + kk];
          if (k0 + KC < DIM_) {
            __builtin_prefetch(qp + KC, 0, 1);
            __builtin_prefetch(sp + KC, 0, 1);
          }
          float4 q4 = *(const float4*)qp;
          float4 s4 = *(const float4*)sp;
          unsigned h0, l0, h1, l1;
          split_pack(q4.x, q4.y, h0, l0);
          split_pack(q4.z, q4.w, h1, l1);
          *(uint2*)&Qhp[0][row * LDSP + (kk >> 1)] = make_uint2(h0, h1);
          *(uint2*)&Qlp[0][row * LDSP + (kk >> 1)] = make_uint2(l0, l1);
          split_pack(s4.x, s4.y, h0, l0);
          split_pack(s4.z, s4.w, h1, l1);
          *(uint2*)&Shp[0][row * LDSP + (kk >> 1)] = make_uint2(h0, h1);
          *(uint2*)&Slp[0][row * LDSP + (kk >> 1)] = make_uint2(l0, l1);
        }
        __syncthreads();
      } else if (MODE == 1) {
        const int kslab = k0 >> 3;                         // uint4 column base
#pragma unroll
        for (int i = 0; i < 2; ++i) {
          int linear = tid + i * 256;                      // 0..511
          int row = linear >> 2;                           // 0..127
          int c4  = linear & 3;                            // uint4 col in slab
          size_t qoff = (size_t)(qbase + row) * (KPROW / 4) + kslab + c4;
          size_t soff = (size_t)(sbase + row) * (KPROW / 4) + kslab + c4;
          if (k0 + KC < DIM_) {
            __builtin_prefetch(&Qh4[qoff + 4], 0, 1);
            __builtin_prefetch(&Sh4[soff + 4], 0, 1);
          }
          *(uint4*)&Qhp[0][row * LDSP + c4 * 4] = Qh4[qoff];
          *(uint4*)&Qlp[0][row * LDSP + c4 * 4] = Ql4[qoff];
          *(uint4*)&Shp[0][row * LDSP + c4 * 4] = Sh4[soff];
          *(uint4*)&Slp[0][row * LDSP + c4 * 4] = Sl4[soff];
        }
        __syncthreads();
      }
#if USE_TDM
      else {
        if (wave < 4) __builtin_amdgcn_s_wait_tensorcnt(0);  // own tile landed
        __syncthreads();                                      // publish to all waves
      }
#endif

      // ---- fragments: pure ds_load_b128 in contiguous k order (k = 16h + s), same for A and B ----
      const unsigned* Qbh = &Qhp[buf][0];
      const unsigned* Qbl = &Qlp[buf][0];
      const unsigned* Sbh = &Shp[buf][0];
      const unsigned* Sbl = &Slp[buf][0];
      Frag ahi[2], alo[2], bhi[4], blo[4];
#pragma unroll
      for (int m = 0; m < 2; ++m) {
        int base = (waveM * 32 + m * 16 + lanelo) * LDSP;
        // A 16x32 bf16: K-pairs {0..7}+8h in v0..7
        set8(ahi[m], *(const uint4*)&Qbh[base + 8 * h], *(const uint4*)&Qbh[base + 8 * h + 4]);
        set8(alo[m], *(const uint4*)&Qbl[base + 8 * h], *(const uint4*)&Qbl[base + 8 * h + 4]);
      }
#pragma unroll
      for (int n = 0; n < 4; ++n) {
        int base = (waveN * 64 + n * 16 + lanelo) * LDSP;
        // B 32x16 bf16: K-pairs {0..7}+8h in v0..7
        set8(bhi[n], *(const uint4*)&Sbh[base + 8 * h], *(const uint4*)&Sbh[base + 8 * h + 4]);
        set8(blo[n], *(const uint4*)&Sbl[base + 8 * h], *(const uint4*)&Sbl[base + 8 * h + 4]);
      }

#if USE_TDM
      if (MODE == 2) {
        if (g + 1 < GSTEPS) tdm_issue(g + 1);   // DMA next slab while we compute
      }
#endif
      if (MODE != 2) __syncthreads();

      // ---- 24 WMMAs: hi*hi + hi*lo + lo*hi (lo*lo ~2^-18, dropped) ----
#pragma unroll
      for (int m = 0; m < 2; ++m)
#pragma unroll
        for (int n = 0; n < 4; ++n) {
          acc[m][n] = __builtin_amdgcn_wmma_f32_16x16x32_bf16(
              false, ahi[m].v, false, bhi[n].v, (short)0, acc[m][n], false, false);
          acc[m][n] = __builtin_amdgcn_wmma_f32_16x16x32_bf16(
              false, ahi[m].v, false, blo[n].v, (short)0, acc[m][n], false, false);
          acc[m][n] = __builtin_amdgcn_wmma_f32_16x16x32_bf16(
              false, alo[m].v, false, bhi[n].v, (short)0, acc[m][n], false, false);
        }
    }

    // ---- running per-lane argmin of  d2' = s2 - 2*qs  ----
#pragma unroll
    for (int m = 0; m < 2; ++m)
#pragma unroll
      for (int r = 0; r < 8; ++r)
#pragma unroll
        for (int n = 0; n < 4; ++n) {
          int loc = st * TS + waveN * 64 + n * 16 + lanelo;
          float d = S2s[loc] - 2.0f * acc[m][n][r];
          if (d < rv[m][r]) { rv[m][r] = d; ri[m][r] = sbase0 + loc; }
        }
  }

  // ---- cross-lane / cross-wave reduction through LDS ----
#pragma unroll
  for (int m = 0; m < 2; ++m)
#pragma unroll
    for (int r = 0; r < 8; ++r) {
      int ql = waveM * 32 + m * 16 + 8 * h + r;     // local query row
      RedV[ql][waveN * 16 + lanelo] = rv[m][r];
      RedI[ql][waveN * 16 + lanelo] = ri[m][r];
    }
  __syncthreads();
  if (tid < TQ) {
    float bv = 3.0e38f; int bi = INT_MAX;
#pragma unroll 4
    for (int j = 0; j < 32; ++j) {
      float v = RedV[tid][j]; int ii = RedI[tid][j];
      if (v < bv || (v == bv && ii < bi)) { bv = v; bi = ii; }
    }
    pv[(size_t)(qbase + tid) * NCHUNK + chunk] = bv;
    pi[(size_t)(qbase + tid) * NCHUNK + chunk] = bi;
  }
}

// ---------------- Kernel 2: final reduce + one-hot output ----------------
__global__ __launch_bounds__(256)
void knn_finalize(const float* __restrict__ pv, const int* __restrict__ pi,
                  const int* __restrict__ slab, float* __restrict__ out) {
  int q = blockIdx.x * 256 + threadIdx.x;
  if (q >= NQ_) return;
  float bv = 3.0e38f; int bi = INT_MAX;
#pragma unroll
  for (int c = 0; c < NCHUNK; ++c) {
    float v = pv[(size_t)q * NCHUNK + c];
    int  ii = pi[(size_t)q * NCHUNK + c];
    if (v < bv || (v == bv && ii < bi)) { bv = v; bi = ii; }
  }
  int lab = slab[bi];
  float4* o = (float4*)&out[(size_t)q * NCLS_];
#pragma unroll
  for (int j = 0; j < 16; ++j) {
    float4 w;
    w.x = (lab == 4 * j + 0) ? 1.f : 0.f;
    w.y = (lab == 4 * j + 1) ? 1.f : 0.f;
    w.z = (lab == 4 * j + 2) ? 1.f : 0.f;
    w.w = (lab == 4 * j + 3) ? 1.f : 0.f;
    o[j] = w;
  }
}

extern "C" void kernel_launch(void* const* d_in, const int* in_sizes, int n_in,
                              void* d_out, int out_size, void* d_ws, size_t ws_size,
                              hipStream_t stream) {
  const float* Sg = (const float*)d_in[0];   // support_embeddings [8192,1024]
  const float* Qg = (const float*)d_in[1];   // query_embeddings   [4096,1024]
  const float* oh = (const float*)d_in[2];   // support_labels_onehot [8192,64]
  float* out = (float*)d_out;                // [4096,64]

  char* ws = (char*)d_ws;
  float* s2g  = (float*)ws;                          // 8192 f32
  int*   slab = (int*)(ws + 32 * 1024);              // 8192 i32
  float* pv   = (float*)(ws + 64 * 1024);            // 4096*8 f32
  int*   pi   = (int*)(ws + 64 * 1024 + 128 * 1024); // 4096*8 i32

  const size_t PACK_OFF = 512 * 1024;
  const size_t QB = (size_t)NQ_ * KPROW * 4;         // 8 MB each (hi/lo)
  const size_t SB = (size_t)NS_ * KPROW * 4;         // 16 MB each (hi/lo)
  unsigned* Qh = (unsigned*)(ws + PACK_OFF);
  unsigned* Ql = (unsigned*)(ws + PACK_OFF + QB);
  unsigned* Sh = (unsigned*)(ws + PACK_OFF + 2 * QB);
  unsigned* Sl = (unsigned*)(ws + PACK_OFF + 2 * QB + SB);
  const bool pre = ws_size >= PACK_OFF + 2 * QB + 2 * SB;

  knn_prep<<<NS_ / 8, 256, 0, stream>>>(Sg, oh, s2g, slab);
  if (pre) {
    knn_split<<<(NQ_ * DIM_ / 4) / 256, 256, 0, stream>>>(Qg, Qh, Ql, NQ_ * DIM_ / 4);
    knn_split<<<(NS_ * DIM_ / 4) / 256, 256, 0, stream>>>(Sg, Sh, Sl, NS_ * DIM_ / 4);
#if USE_TDM
    knn_gemm_argmin<2><<<dim3(NQ_ / TQ, NCHUNK), 256, 0, stream>>>(
        Sg, Qg, (const uint4*)Qh, (const uint4*)Ql, (const uint4*)Sh, (const uint4*)Sl,
        s2g, pv, pi);
#else
    knn_gemm_argmin<1><<<dim3(NQ_ / TQ, NCHUNK), 256, 0, stream>>>(
        Sg, Qg, (const uint4*)Qh, (const uint4*)Ql, (const uint4*)Sh, (const uint4*)Sl,
        s2g, pv, pi);
#endif
  } else {
    knn_gemm_argmin<0><<<dim3(NQ_ / TQ, NCHUNK), 256, 0, stream>>>(
        Sg, Qg, nullptr, nullptr, nullptr, nullptr, s2g, pv, pi);
  }
  knn_finalize<<<(NQ_ + 255) / 256, 256, 0, stream>>>(pv, pi, slab, out);
}
